// Cell_A_10746008174946
// MI455X (gfx1250) — matrix operand order repaired and checked by reading, not compiled, not run on hardware
//
#include <hip/hip_runtime.h>

// ---------------- problem constants (match reference) ----------------
#define BATCH 8
#define ICH   64
#define LEN   1024
#define OC1   256      // CP*AP
#define G2K   9
#define PAD1  4
#define LT    32       // L-tile per workgroup in kernel 1
#define KTOT  576      // ICH*G2K
#define KST   32       // WMMA K step (bf16)
#define NKS   18       // KTOT/KST
#define WSLICE (OC1 * KST)   // 8192 elements per weight K-slice
#define XCOLS (LT + 8)       // x tile width incl. halo

#define CPn   32
#define APn   8
#define OC2   256      // CSA*ASA
#define K2    24       // G3K*APn (padded to 32 for WMMA)
#define CSAn  16
#define ASAn  16

typedef __attribute__((ext_vector_type(16))) __bf16        v16bf;
typedef __attribute__((ext_vector_type(8)))  float         v8f;
typedef __attribute__((ext_vector_type(4)))  unsigned int  v4u;
typedef __attribute__((ext_vector_type(2)))  unsigned int  v2u;
typedef __attribute__((ext_vector_type(4)))  int           v4i;
typedef __attribute__((ext_vector_type(8)))  int           v8i;

union Frag { v16bf v; v4u q[2]; };
static_assert(sizeof(v16bf) == 32, "v16bf must be 32B");

// fp32 -> bf16 (round-to-nearest-even) as raw bits
__device__ __forceinline__ unsigned short f2bf_bits(float f) {
  unsigned int u = __builtin_bit_cast(unsigned int, f);
  u += 0x7FFFu + ((u >> 16) & 1u);
  return (unsigned short)(u >> 16);
}

// ---------------------------------------------------------------------
// TDM: 1-D tensor_load_to_lds of `nelem` bf16 elements (wave-level op;
// call from a single wave). D# per CDNA5 ISA.
// ---------------------------------------------------------------------
__device__ __forceinline__ void tdm_load_bf16(unsigned short* lds_dst,
                                              const unsigned short* gsrc,
                                              int nelem) {
  // LDS byte offset of destination (addrspace(3) ptr -> 32-bit offset)
  unsigned lds_off = (unsigned)(unsigned long long)
      (__attribute__((address_space(3))) unsigned short*)(void*)lds_dst;
  unsigned long long ga = (unsigned long long)gsrc;

  v4u g0;
  g0[0] = 1u;                                   // count=1 (valid descriptor)
  g0[1] = lds_off;                              // lds_addr
  g0[2] = (unsigned)ga;                         // global_addr[31:0]
  g0[3] = (unsigned)((ga >> 32) & 0x1FFFFFFull) // global_addr[56:32]
          | (2u << 30);                         // type = 2 ("image")

  v8i g1;
  g1[0] = 1 << 16;                              // data_size=1 (2B); no mask
  g1[1] = (int)((nelem & 0xFFFF) << 16);        // tensor_dim0[15:0]
  g1[2] = (int)(((unsigned)nelem >> 16) & 0xFFFFu) | (1 << 16); // dim0 hi | tensor_dim1=1
  g1[3] = (int)((nelem & 0xFFFF) << 16);        // tile_dim0 = nelem
  g1[4] = 1;                                    // tile_dim1 = 1, tile_dim2 = 0
  g1[5] = nelem;                                // tensor_dim0_stride lo32
  g1[6] = 0;                                    // stride hi | dim1_stride lo
  g1[7] = 0;

  v4i z4 = {0, 0, 0, 0};
#if defined(__clang_major__) && (__clang_major__ >= 23)
  v8i z8 = {0, 0, 0, 0, 0, 0, 0, 0};
  __builtin_amdgcn_tensor_load_to_lds(g0, g1, z4, z4, z8, 0);
#else
  __builtin_amdgcn_tensor_load_to_lds(g0, g1, z4, z4, 0);
#endif
}

// =====================================================================
// Kernel 0: one-time weight packing (fp32 -> bf16, WMMA-B swizzled)
//  w1bf [ks][oc][kk]  (18 x 16KB contiguous K-slices)
//  w2bf [oc][kk]      (K padded 24->32 with zeros)
// =====================================================================
__global__ __launch_bounds__(256) void pack_weights(
    const float* __restrict__ w1, const float* __restrict__ w2,
    unsigned short* __restrict__ w1bf, unsigned short* __restrict__ w2bf) {
  int e = blockIdx.x * 256 + threadIdx.x;
  if (e < KTOT * OC1) {                 // 147456
    int ks = e >> 13;                   // / 8192
    int r  = e & 8191;
    int oc = r >> 5, kk = r & 31;
    w1bf[e] = f2bf_bits(w1[oc * KTOT + ks * KST + kk]);
  }
  if (e < OC2 * KST) {                  // 8192
    int oc = e >> 5, kk = e & 31;
    w2bf[e] = (kk < K2) ? f2bf_bits(w2[oc * K2 + kk]) : (unsigned short)0;
  }
}

// =====================================================================
// Kernel 1: conv1 (implicit GEMM, bf16 WMMA) + bias + squash -> y (bf16)
// grid (LEN/LT, BATCH), 256 threads (8 waves)
// - x tile is converted to bf16 ONCE at staging; the per-K-step A panel
//   build is a pure branch-free gather: 4x ds_load_u16 + 1x ds_store_b64
//   per thread (no conversion VALU inside the K loop).
// - Weight K-slices arrive via double-buffered TDM tensor_load_to_lds,
//   issued by wave 0 one slice ahead of compute (TENSORcnt pipelining).
// =====================================================================
__global__ __launch_bounds__(256) void conv1_squash_wmma(
    const float* __restrict__ x, const unsigned short* __restrict__ w1bf,
    const float* __restrict__ b1, unsigned short* __restrict__ y_ws) {
  __shared__ unsigned short lds_xh[ICH * XCOLS];                 // bf16 x, 5.1 KB
  __shared__ unsigned short ktab[KTOT];                          // 1.13 KB
  __shared__ alignas(16) unsigned short aslT[LT * KST];          // [m][kk] 2 KB
  // 32 KB region: double-buffered weight panel during the K loop,
  // reused as the fp32 y tile afterwards (never live simultaneously).
  __shared__ alignas(16) unsigned char smem_u[2 * WSLICE * sizeof(unsigned short)];
  unsigned short* wslT  = reinterpret_cast<unsigned short*>(smem_u);
  float*          ytile = reinterpret_cast<float*>(smem_u);

  const int tid  = threadIdx.x;
  const int lane = tid & 31;
  const int wid  = tid >> 5;
  const int b    = blockIdx.y;
  const int l0   = blockIdx.x * LT;

  // kick off TDM for slice 0 while everyone stages x
  if (wid == 0) tdm_load_bf16(wslT, w1bf, WSLICE);

  // stage x tile with zero-padded halo, converting to bf16 once
  for (int e = tid; e < ICH * XCOLS; e += 256) {
    int ic = e / XCOLS;
    int c  = e - ic * XCOLS;
    int lg = l0 + c - PAD1;
    lds_xh[e] = (lg >= 0 && lg < LEN) ? f2bf_bits(x[(b * ICH + ic) * LEN + lg])
                                      : (unsigned short)0;
  }
  // K -> (ic,t) offset table
  for (int k = tid; k < KTOT; k += 256) {
    int ic = k / G2K;
    int t  = k - ic * G2K;
    ktab[k] = (unsigned short)(ic * XCOLS + t);
  }

  const int m_sub  = wid >> 2;            // 0..1
  const int n_base = (wid & 3) * 4;       // 0,4,8,12 (x16 cols)
  const int mrow   = m_sub * 16 + (lane & 15);
  const int kbA    = (lane >> 4) << 3;    // A: lanes 16-31 hold +8 K half
  const int kbB    = (lane >> 4) << 3;    // B: lanes 16-31 hold +8 K half
  const int ncol   = lane & 15;

  // branch-free A-stage mapping: thread -> (m, 4 consecutive kk)
  const int sm  = tid >> 3;        // 0..31 (m row)
  const int sk4 = (tid & 7) * 4;   // 0,4,...,28

  v8f acc0 = {0.f,0.f,0.f,0.f,0.f,0.f,0.f,0.f};
  v8f acc1 = acc0, acc2 = acc0, acc3 = acc0;

  for (int ks = 0; ks < NKS; ++ks) {
    __syncthreads();   // prev iter's compute (reads of old buffers) done

    // stage A panel [m][kk]: pure bf16 gather, one b64 store per thread
    {
      const int kg = ks * KST + sk4;
      unsigned int t0 = lds_xh[ktab[kg + 0] + sm];
      unsigned int t1 = lds_xh[ktab[kg + 1] + sm];
      unsigned int t2 = lds_xh[ktab[kg + 2] + sm];
      unsigned int t3 = lds_xh[ktab[kg + 3] + sm];
      v2u pk = { t0 | (t1 << 16), t2 | (t3 << 16) };
      *(v2u*)&aslT[sm * KST + sk4] = pk;
    }

    if (wid == 0) {
      __builtin_amdgcn_s_wait_tensorcnt(0);       // slice ks has landed
      if (ks + 1 < NKS)                           // prefetch slice ks+1
        tdm_load_bf16(wslT + ((ks + 1) & 1) * WSLICE,
                      w1bf + (size_t)(ks + 1) * WSLICE, WSLICE);
    }
    __syncthreads();   // aslT ready; TDM data visible to all waves

    const unsigned short* wbase = wslT + (ks & 1) * WSLICE;

    // A fragment: two aligned 16B runs (K kb..kb+7, kb+16..kb+23)
    Frag fa;
    const unsigned short* abase = &aslT[mrow * KST];
    fa.q[0] = *(const v4u*)(abase + kbA);
    fa.q[1] = *(const v4u*)(abase + kbA + 16);

    #pragma unroll
    for (int n = 0; n < 4; ++n) {
      int oc = (n_base + n) * 16 + ncol;
      Frag fb;
      const unsigned short* bbase = wbase + oc * KST + kbB;
      fb.q[0] = *(const v4u*)(bbase);
      fb.q[1] = *(const v4u*)(bbase + 16);
      v8f& acc = (n == 0 ? acc0 : n == 1 ? acc1 : n == 2 ? acc2 : acc3);
      acc = __builtin_amdgcn_wmma_f32_16x16x32_bf16(
          false, fa.v, false, fb.v, (short)0, acc, false, false);
    }
  }

  __syncthreads();   // last slice consumed; smem_u becomes ytile
  // bias + stash raw y in LDS (D layout: lanes16-31 rows +8)
  #pragma unroll
  for (int n = 0; n < 4; ++n) {
    v8f acc = (n == 0 ? acc0 : n == 1 ? acc1 : n == 2 ? acc2 : acc3);
    int oc = (n_base + n) * 16 + ncol;
    #pragma unroll
    for (int r = 0; r < 8; ++r) {
      int row = m_sub * 16 + r + ((lane >> 4) << 3);
      ytile[row * OC1 + oc] = acc[r] + b1[oc];
    }
  }
  __syncthreads();

  // squash over ap=8 per (l, cp); one packed 16B bf16 store per group
  for (int g = tid; g < LT * CPn; g += 256) {
    int l = g >> 5, cp = g & 31;
    const float* yr = &ytile[l * OC1 + cp * APn];
    float sq = 0.f;
    #pragma unroll
    for (int a = 0; a < APn; ++a) sq += yr[a] * yr[a];
    float sc = sq / (1.f + sq) * rsqrtf(sq + 1e-8f);
    v4u pk;
    #pragma unroll
    for (int a2 = 0; a2 < 4; ++a2) {
      unsigned int lo = f2bf_bits(yr[2 * a2]     * sc);
      unsigned int hi = f2bf_bits(yr[2 * a2 + 1] * sc);
      pk[a2] = lo | (hi << 16);
    }
    *(v4u*)&y_ws[((b * LEN) + l0 + l) * OC1 + cp * APn] = pk;
  }
}

// =====================================================================
// Kernel 2: conv2 (one WMMA K-step, K=24 padded to 32) + dynamic routing
// grid (LEN, BATCH), 256 threads; V[32cp x 256oc] lives entirely in LDS.
// w2 panel arrives via one TDM tensor_load_to_lds.
// =====================================================================
__global__ __launch_bounds__(256) void conv2_routing_wmma(
    const unsigned short* __restrict__ y_ws, const unsigned short* __restrict__ w2bf,
    const float* __restrict__ b2, float* __restrict__ out) {
  __shared__ alignas(16) unsigned short y3[3 * OC1];    // y rows l-1..l+1
  __shared__ alignas(16) unsigned short w2t[OC2 * KST]; // [oc][kk] (16 KB)
  __shared__ float Vl[CPn * OC2];                       // V for this l (32 KB)
  __shared__ float br[CPn * CSAn];
  __shared__ float cc[CPn * CSAn];
  __shared__ float sv[CSAn * ASAn];
  __shared__ float vv[CSAn * ASAn];
  __shared__ float scl[CSAn];

  const int tid  = threadIdx.x;
  const int lane = tid & 31;
  const int wid  = tid >> 5;
  const int b    = blockIdx.y;
  const int l    = blockIdx.x;

  // wave 0: async-load the packed w2 panel straight into LDS
  if (wid == 0) tdm_load_bf16(w2t, w2bf, OC2 * KST);

  // stage the 3 y rows (zero-pad at sequence edges)
  for (int e = tid; e < 3 * OC1; e += 256) {
    int t = e >> 8, oc = e & 255;
    int lg = l + t - 1;
    y3[e] = (lg >= 0 && lg < LEN) ? y_ws[(b * LEN + lg) * OC1 + oc]
                                  : (unsigned short)0;
  }
  if (wid == 0) __builtin_amdgcn_s_wait_tensorcnt(0);
  __syncthreads();

  // ---- conv2 WMMA: M = cp (32 rows), N = oc2 (256), K = 32 ----
  const int m_sub  = wid >> 2;
  const int n_base = (wid & 3) * 4;
  const int cp     = m_sub * 16 + (lane & 15);
  const int kbA    = (lane >> 4) << 3;
  const int kbB    = (lane >> 4) << 3;
  const int ncol   = lane & 15;

  // A fragment: K-runs are contiguous in y3 (t fixed per run, a = 0..7)
  Frag fa;
  const int t1 = kbA >> 3;                     // 0 or 1
  fa.q[0] = *(const v4u*)&y3[t1 * OC1 + cp * APn];
  v4u zero4 = {0u, 0u, 0u, 0u};
  fa.q[1] = (kbA == 0) ? *(const v4u*)&y3[2 * OC1 + cp * APn] : zero4;

  #pragma unroll
  for (int n = 0; n < 4; ++n) {
    int oc = (n_base + n) * 16 + ncol;
    Frag fb;
    const unsigned short* bbase = &w2t[oc * KST + kbB];
    fb.q[0] = *(const v4u*)(bbase);
    fb.q[1] = *(const v4u*)(bbase + 16);
    v8f acc = {0.f,0.f,0.f,0.f,0.f,0.f,0.f,0.f};
    acc = __builtin_amdgcn_wmma_f32_16x16x32_bf16(
        false, fa.v, false, fb.v, (short)0, acc, false, false);
    #pragma unroll
    for (int r = 0; r < 8; ++r) {
      int row = m_sub * 16 + r + ((lane >> 4) << 3);
      Vl[row * OC2 + oc] = acc[r] + b2[oc];
    }
  }
  __syncthreads();

  // ---- dynamic routing (3 iterations) ----
  for (int e = tid; e < CPn * CSAn; e += 256) br[e] = 0.f;
  __syncthreads();

  const int i = tid >> 4;   // cSA
  const int j = tid & 15;   // aSA
  float v_ij = 0.f;

  for (int it = 0; it < 3; ++it) {
    if (tid < CPn) {        // softmax over cSA per cp
      int row = tid * CSAn;
      float mx = -1e30f;
      #pragma unroll
      for (int k = 0; k < CSAn; ++k) mx = fmaxf(mx, br[row + k]);
      float sum = 0.f;
      #pragma unroll
      for (int k = 0; k < CSAn; ++k) {
        float e = __expf(br[row + k] - mx);
        cc[row + k] = e;
        sum += e;
      }
      float inv = 1.f / sum;
      #pragma unroll
      for (int k = 0; k < CSAn; ++k) cc[row + k] *= inv;
    }
    __syncthreads();

    float s = 0.f;          // s[i][j] = sum_cp c[cp][i] * V[cp][i][j]
    #pragma unroll 4
    for (int p = 0; p < CPn; ++p) s += cc[p * CSAn + i] * Vl[p * OC2 + i * ASAn + j];
    sv[i * ASAn + j] = s;
    __syncthreads();

    if (tid < CSAn) {       // squash scale per i
      float sq = 0.f;
      #pragma unroll
      for (int k = 0; k < ASAn; ++k) {
        float t = sv[tid * ASAn + k];
        sq += t * t;
      }
      scl[tid] = sq / (1.f + sq) * rsqrtf(sq + 1e-8f);
    }
    __syncthreads();

    v_ij = sv[i * ASAn + j] * scl[i];
    vv[i * ASAn + j] = v_ij;
    __syncthreads();

    if (it < 2) {           // b += a, a[cp][i] = sum_j V[cp][i][j]*v[i][j]
      for (int e = tid; e < CPn * CSAn; e += 256) {
        int p = e >> 4, ii = e & 15;
        float a = 0.f;
        #pragma unroll
        for (int k = 0; k < ASAn; ++k) a += Vl[p * OC2 + ii * ASAn + k] * vv[ii * ASAn + k];
        br[e] += a;
      }
      __syncthreads();
    }
  }

  // out: [B, L*CSA, ASA]
  out[((b * LEN + l) * CSAn + i) * ASAn + j] = v_ij;
}

// =====================================================================
extern "C" void kernel_launch(void* const* d_in, const int* in_sizes, int n_in,
                              void* d_out, int out_size, void* d_ws, size_t ws_size,
                              hipStream_t stream) {
  const float* x  = (const float*)d_in[0];   // [8,64,1024]
  const float* w1 = (const float*)d_in[1];   // [256,64,9]
  const float* b1 = (const float*)d_in[2];   // [256]
  const float* w2 = (const float*)d_in[3];   // [256,1,3,8]
  const float* b2 = (const float*)d_in[4];   // [256]
  float* out = (float*)d_out;                // [8, 16384, 16]

  // workspace layout (bf16): y [B*L*256] = 4MB | w1bf 288KB | w2bf 16KB
  unsigned short* y_ws = (unsigned short*)d_ws;
  unsigned short* w1bf = y_ws + (size_t)BATCH * LEN * OC1;
  unsigned short* w2bf = w1bf + (size_t)KTOT * OC1;

  pack_weights<<<dim3((KTOT * OC1 + 255) / 256), 256, 0, stream>>>(w1, w2, w1bf, w2bf);

  dim3 g1(LEN / LT, BATCH);                  // (32, 8)
  conv1_squash_wmma<<<g1, 256, 0, stream>>>(x, w1bf, b1, y_ws);

  dim3 g2(LEN, BATCH);                       // (1024, 8)
  conv2_routing_wmma<<<g2, 256, 0, stream>>>(y_ws, w2bf, b2, out);
}
